// GATLayer_5403068859082
// MI455X (gfx1250) — matrix operand order repaired and checked by reading, not compiled, not run on hardware
//
#include <hip/hip_runtime.h>

#define NN 10000
#define NE 640000
#define D  128

typedef __attribute__((ext_vector_type(16))) __bf16          v16bf;
typedef __attribute__((ext_vector_type(16))) unsigned short  v16us;
typedef __attribute__((ext_vector_type(8)))  float           v8f;
typedef __attribute__((ext_vector_type(4)))  float           v4f;

union BF16x16 { v16us u; v16bf b; };

// native hardware f32 -> bf16 convert (RTNE), bit pattern for storing to ws
__device__ __forceinline__ unsigned short f2bf_bits(float f) {
    union { __bf16 h; unsigned short s; } c;
    c.h = (__bf16)f;
    return c.s;
}

__device__ __forceinline__ void atomicMaxF(float* addr, float v) {
    if (v >= 0.0f) atomicMax((int*)addr, __float_as_int(v));
    else           atomicMin((unsigned int*)addr, __float_as_uint(v));
}

// ---------------- weight pre-swizzle into WMMA B-layout (bf16) --------------
// B layout (32x16, bf16): lane L holds column N=L&15; element j <-> K = 8*(L>>4)+(j<8 ? j : j+8)
__global__ void pack_w(const float* __restrict__ Wlin, const float* __restrict__ Wapp,
                       unsigned short* wA, unsigned short* wB, unsigned short* wApp) {
    int t = blockIdx.x * 256 + threadIdx.x;
    const float* W; unsigned short* dp; int koff, CH, u;
    if      (t < 16384) { u = t;          W = Wlin; dp = wA;   koff = 0;   CH = 4; }
    else if (t < 32768) { u = t - 16384;  W = Wlin; dp = wB;   koff = 128; CH = 4; }
    else if (t < 65536) { u = t - 32768;  W = Wapp; dp = wApp; koff = 0;   CH = 8; }
    else return;
    int j    = u & 15;
    int lane = (u >> 4) & 31;
    int tc   = u >> 9;              // ntile*CH + chunk
    int chunk = tc % CH, ntile = tc / CH;
    int kh = lane >> 4, nl = lane & 15;
    int kg  = chunk * 32 + 8 * kh + (j < 8 ? j : j + 8) + koff;
    int row = ntile * 16 + nl;
    dp[u] = f2bf_bits(W[row * 256 + kg]);
}

// ---------------- A-operand loader: 16 rows x 128 cols, chunk c -------------
// A layout (16x32 bf16): lane L = row M=L&15; elem j <-> K = c*32 + 8*(L>>4) + (j<8 ? j : j+8)
__device__ __forceinline__ v16bf load_A_chunk(const float* __restrict__ row, int c, int kh) {
    float fa[16];
    const float* p0 = row + c * 32 + kh * 8;
    *(v4f*)&fa[0]  = *(const v4f*)(p0);
    *(v4f*)&fa[4]  = *(const v4f*)(p0 + 4);
    *(v4f*)&fa[8]  = *(const v4f*)(p0 + 16);
    *(v4f*)&fa[12] = *(const v4f*)(p0 + 20);
    v16bf a;
#pragma unroll
    for (int j = 0; j < 16; j++) a[j] = (__bf16)fa[j];   // native v_cvt*bf16_f32
    return a;
}

// ---------------- per-node attention scalars --------------------------------
__global__ void node_attn(const float* __restrict__ nf, const float* __restrict__ Wattn,
                          float* a_src, float* a_dst) {
    int gt = blockIdx.x * 256 + threadIdx.x;
    int wave = gt >> 5, lane = gt & 31;
    if (wave >= NN) return;
    const float* row = nf + wave * D;
    float s0 = 0.f, s1 = 0.f;
#pragma unroll
    for (int i = 0; i < 4; i++) {
        float f = row[lane * 4 + i];
        s0 += f * Wattn[lane * 4 + i];
        s1 += f * Wattn[128 + lane * 4 + i];
    }
#pragma unroll
    for (int off = 16; off > 0; off >>= 1) {
        s0 += __shfl_down(s0, off, 32);
        s1 += __shfl_down(s1, off, 32);
    }
    if (lane == 0) { a_src[wave] = s0; a_dst[wave] = s1; }
}

// ---------------- ms = nfeats @ W_linA^T  (WMMA bf16) -----------------------
__global__ void __launch_bounds__(256) node_ms(const float* __restrict__ nf,
                                               const unsigned short* __restrict__ wA,
                                               float* __restrict__ ms) {
    int wave = (blockIdx.x * 256 + threadIdx.x) >> 5;
    int lane = threadIdx.x & 31;
    if (wave >= NN / 16) return;
    int nbase = wave << 4;
    int nl = lane & 15, kh = lane >> 4;
    v16bf A[4];
    const float* arow = nf + (nbase + nl) * D;
#pragma unroll
    for (int c = 0; c < 4; c++) A[c] = load_A_chunk(arow, c, kh);
    for (int nt = 0; nt < 8; nt++) {
        v8f acc = {0.f,0.f,0.f,0.f,0.f,0.f,0.f,0.f};
#pragma unroll
        for (int c = 0; c < 4; c++) {
            BF16x16 B;
            B.u = *(const v16us*)(wA + (((nt * 4 + c) * 32 + lane) << 4));
            acc = __builtin_amdgcn_wmma_f32_16x16x32_bf16(false, A[c], false, B.b,
                                                          (short)0, acc, false, false);
        }
        int col = nt * 16 + nl;
#pragma unroll
        for (int r = 0; r < 8; r++)
            ms[(nbase + r + 8 * kh) * D + col] = acc[r];
    }
}

// ---------------- init emax/denom/z -----------------------------------------
__global__ void init_ws(float* emax, float* denom, float* z) {
    int t = blockIdx.x * 256 + threadIdx.x;
    if (t < NN) { ((unsigned*)emax)[t] = 0xFF800000u; denom[t] = 0.f; }
    if (t < NN * D) z[t] = 0.f;
}

// ---------------- edge logits + segment max ---------------------------------
__global__ void edge_e(const int* __restrict__ src, const int* __restrict__ dst,
                       const float* __restrict__ a_src, const float* __restrict__ a_dst,
                       const float* __restrict__ battn, float* eArr, float* emax) {
    int i = blockIdx.x * 256 + threadIdx.x;
    if (i >= NE) return;
    int s = src[i], d = dst[i];
    float v = a_src[s] + a_dst[d] + battn[0];
    v = v > 0.f ? v : 0.01f * v;          // leaky_relu(0.01)
    eArr[i] = v;
    atomicMaxF(&emax[d], v);
}

// ---------------- exp + segment sum -----------------------------------------
__global__ void edge_ex(const int* __restrict__ dst, const float* __restrict__ eArr,
                        const float* __restrict__ emax, float* exArr, float* denom) {
    int i = blockIdx.x * 256 + threadIdx.x;
    if (i >= NE) return;
    int d = dst[i];
    float v = __expf(eArr[i] - emax[d]);
    exArr[i] = v;
    atomicAdd(&denom[d], v);
}

// ---------------- main: ef@W_linB^T (WMMA) + alpha*( . + ms[src]) scatter ---
__global__ void __launch_bounds__(256) edge_gemm(const float* __restrict__ ef,
                                                 const int* __restrict__ src,
                                                 const int* __restrict__ dst,
                                                 const float* __restrict__ exArr,
                                                 const float* __restrict__ denom,
                                                 const float* __restrict__ ms,
                                                 const unsigned short* __restrict__ wB,
                                                 float* __restrict__ z) {
    int wave = (blockIdx.x * 256 + threadIdx.x) >> 5;      // 40000 waves, exact
    int lane = threadIdx.x & 31;
    int ebase = wave << 4;
    int nl = lane & 15, kh = lane >> 4;

    v16bf A[4];
    const float* arow = ef + (ebase + nl) * D;
#pragma unroll
    for (int c = 0; c < 4; c++) A[c] = load_A_chunk(arow, c, kh);

    // per-edge scalars for the 8 D-rows this lane owns (row r + 8*kh)
    float alpha8[8]; int s8[8], d8[8];
#pragma unroll
    for (int r = 0; r < 8; r++) {
        int e = ebase + r + 8 * kh;
        int s = src[e], d = dst[e];
        alpha8[r] = exArr[e] / denom[d];
        s8[r] = s * D; d8[r] = d * D;
    }

    for (int nt = 0; nt < 8; nt++) {
        v8f acc = {0.f,0.f,0.f,0.f,0.f,0.f,0.f,0.f};
#pragma unroll
        for (int c = 0; c < 4; c++) {
            BF16x16 B;
            B.u = *(const v16us*)(wB + (((nt * 4 + c) * 32 + lane) << 4));
            acc = __builtin_amdgcn_wmma_f32_16x16x32_bf16(false, A[c], false, B.b,
                                                          (short)0, acc, false, false);
        }
        int col = nt * 16 + nl;
#pragma unroll
        for (int r = 0; r < 8; r++) {
            float val = (acc[r] + ms[s8[r] + col]) * alpha8[r];
            atomicAdd(&z[d8[r] + col], val);      // f32 atomic, resolves in L2
        }
    }
}

// ---------------- out = relu([nfeats,z] @ W_app^T + b) (WMMA, K=256) --------
__global__ void __launch_bounds__(256) out_gemm(const float* __restrict__ nf,
                                                const float* __restrict__ z,
                                                const unsigned short* __restrict__ wApp,
                                                const float* __restrict__ bapp,
                                                float* __restrict__ out) {
    int wave = (blockIdx.x * 256 + threadIdx.x) >> 5;
    int lane = threadIdx.x & 31;
    if (wave >= NN / 16) return;
    int nbase = wave << 4;
    int nl = lane & 15, kh = lane >> 4;
    v16bf A[8];
    const float* r0 = nf + (nbase + nl) * D;
    const float* r1 = z  + (nbase + nl) * D;
#pragma unroll
    for (int c = 0; c < 8; c++)
        A[c] = load_A_chunk((c < 4) ? r0 : r1, c & 3, kh);
    for (int nt = 0; nt < 8; nt++) {
        v8f acc = {0.f,0.f,0.f,0.f,0.f,0.f,0.f,0.f};
#pragma unroll
        for (int c = 0; c < 8; c++) {
            BF16x16 B;
            B.u = *(const v16us*)(wApp + (((nt * 8 + c) * 32 + lane) << 4));
            acc = __builtin_amdgcn_wmma_f32_16x16x32_bf16(false, A[c], false, B.b,
                                                          (short)0, acc, false, false);
        }
        int col = nt * 16 + nl;
        float b = bapp[col];
#pragma unroll
        for (int r = 0; r < 8; r++) {
            float v = acc[r] + b;
            out[(nbase + r + 8 * kh) * D + col] = v > 0.f ? v : 0.f;
        }
    }
}

extern "C" void kernel_launch(void* const* d_in, const int* in_sizes, int n_in,
                              void* d_out, int out_size, void* d_ws, size_t ws_size,
                              hipStream_t stream) {
    const float* nf    = (const float*)d_in[0];
    const float* ef    = (const float*)d_in[1];
    const int*   src   = (const int*)  d_in[2];
    const int*   dst   = (const int*)  d_in[3];
    const float* Wlin  = (const float*)d_in[4];
    const float* Wapp  = (const float*)d_in[5];
    const float* bapp  = (const float*)d_in[6];
    const float* Wattn = (const float*)d_in[7];
    const float* battn = (const float*)d_in[8];
    float* out = (float*)d_out;
    char*  ws  = (char*)d_ws;

    unsigned short* wA    = (unsigned short*)(ws + 0);        //  32 KB
    unsigned short* wB    = (unsigned short*)(ws + 32768);    //  32 KB
    unsigned short* wAp   = (unsigned short*)(ws + 65536);    //  64 KB
    float* a_src = (float*)(ws + 131072);                     //  40 KB (padded)
    float* a_dst = (float*)(ws + 172032);
    float* msbuf = (float*)(ws + 212992);                     // 5.12 MB
    float* eArr  = (float*)(ws + 5332992);                    // 2.56 MB
    float* exArr = (float*)(ws + 7892992);                    // 2.56 MB
    float* emax  = (float*)(ws + 10452992);
    float* denom = (float*)(ws + 10492992);
    float* z     = (float*)(ws + 10532992);                   // 5.12 MB -> ~14.9 MB total

    pack_w   <<<256,  256, 0, stream>>>(Wlin, Wapp, wA, wB, wAp);
    node_attn<<<1250, 256, 0, stream>>>(nf, Wattn, a_src, a_dst);
    node_ms  <<<79,   256, 0, stream>>>(nf, wA, msbuf);
    init_ws  <<<5000, 256, 0, stream>>>(emax, denom, z);
    edge_e   <<<2500, 256, 0, stream>>>(src, dst, a_src, a_dst, battn, eArr, emax);
    edge_ex  <<<2500, 256, 0, stream>>>(dst, eArr, emax, exArr, denom);
    edge_gemm<<<5000, 256, 0, stream>>>(ef, src, dst, exArr, denom, msbuf, wB, z);
    out_gemm <<<79,   256, 0, stream>>>(nf, z, wAp, bapp, out);
}
